// Attention_18056042512624
// MI455X (gfx1250) — matrix operand order repaired and checked by reading, not compiled, not run on hardware
//
#include <hip/hip_runtime.h>
#include <hip/hip_bf16.h>

// ---------------------------------------------------------------------------
// Problem constants (from reference): B=128, N=197, C=768, H=12, hd=64,
// SPARSITY=0.5 -> kth=98 (keep >= 99th-largest), num_prop=16, num_kept=181.
// Padded shapes: NP=208 (rows, 13 WMMA tiles), KP=224 (K dim, 7 x 32).
// ---------------------------------------------------------------------------
#define BB 128
#define NN 197
#define CC 768
#define HH 12
#define HD 64
#define NP 208
#define KP 224
#define NKEPT 181
#define NPROP 16
#define ROWS (BB * NN)          // 25216 = 1576 * 16
#define MT_QKV 1576
#define NT64_QKV 36             // 2304 / 64
#define ROWS_P (BB * NKEPT)     // 23168 = 1448 * 16
#define MT_PROJ 1448
#define NT64_PROJ 12            // 768 / 64

typedef __attribute__((ext_vector_type(16))) __bf16 bf16x16;
typedef __attribute__((ext_vector_type(8)))  float  f32x8;
typedef __attribute__((ext_vector_type(4)))  unsigned int uint32x4;

union BFV { struct { uint32x4 lo; uint32x4 hi; } u; bf16x16 v; };

// A-operand (16x32 bf16, M across lanes%16): lane half holds contiguous K block, kb=(lane&16)?16:0
__device__ __forceinline__ bf16x16 load_opA(const __bf16* p) {
    BFV t;
    t.u.lo = *(const uint32x4*)(p);
    t.u.hi = *(const uint32x4*)(p + 8);
    return t.v;
}
// B-operand (32x16 bf16, N across lanes%16): lane half holds contiguous K block, kb2=(lane&16)?16:0
__device__ __forceinline__ bf16x16 load_opB(const __bf16* p) {
    BFV t;
    t.u.lo = *(const uint32x4*)(p);
    t.u.hi = *(const uint32x4*)(p + 8);
    return t.v;
}

__device__ __forceinline__ f32x8 wmma_bf16(bf16x16 a, bf16x16 b, f32x8 c) {
    return __builtin_amdgcn_wmma_f32_16x16x32_bf16(false, a, false, b, (short)0, c, false, false);
}

__device__ __forceinline__ float wsum(float v) {
#pragma unroll
    for (int o = 16; o > 0; o >>= 1) v += __shfl_xor(v, o, 32);
    return v;
}
__device__ __forceinline__ float wmaxf(float v) {
#pragma unroll
    for (int o = 16; o > 0; o >>= 1) v = fmaxf(v, __shfl_xor(v, o, 32));
    return v;
}

// ---------------------------------------------------------------------------
// K0: f32 -> bf16 cast
// ---------------------------------------------------------------------------
__global__ void __launch_bounds__(256) cast_bf16_kernel(const float* __restrict__ in,
                                                        __bf16* __restrict__ out, int n) {
    int i = blockIdx.x * 256 + threadIdx.x;
    if (i < n) out[i] = (__bf16)in[i];
}

// ---------------------------------------------------------------------------
// K1: QKV GEMM  [25216 x 768] @ Wqkv^T(768 x 2304) + bias.
// One wave computes a 16x64 tile (4 accumulators). Software-pipelined:
// operands for K-step ks+1 are in flight while WMMAs for ks execute, so the
// per-step wait is partial (s_wait_loadcnt > 0) instead of a full stall.
// Epilogue scatters into q (scaled 0.125), k ([bh,NP,64]) and vT ([bh,64,KP]).
// ---------------------------------------------------------------------------
__global__ void __launch_bounds__(256) qkv_gemm_kernel(const __bf16* __restrict__ xb,
                                                       const __bf16* __restrict__ wb,
                                                       const float* __restrict__ bias,
                                                       __bf16* __restrict__ qb,
                                                       __bf16* __restrict__ kbuf,
                                                       __bf16* __restrict__ vtb) {
    int wid = (blockIdx.x * 256 + threadIdx.x) >> 5;
    if (wid >= MT_QKV * NT64_QKV) return;
    int lane = threadIdx.x & 31;
    int mt = wid / NT64_QKV, nt = wid % NT64_QKV;
    int kbA = (lane & 16) ? 16 : 0;
    int kbB = (lane & 16) ? 16 : 0;
    int hi8 = (lane & 16) ? 8 : 0;
    const __bf16* arow = xb + (size_t)(mt * 16 + (lane & 15)) * CC;
    const __bf16* brow[4];
#pragma unroll
    for (int s = 0; s < 4; ++s)
        brow[s] = wb + (size_t)(nt * 64 + s * 16 + (lane & 15)) * CC;

    f32x8 acc[4] = {{}, {}, {}, {}};
    bf16x16 a = load_opA(arow + kbA);
    bf16x16 bm[4];
#pragma unroll
    for (int s = 0; s < 4; ++s) bm[s] = load_opB(brow[s] + kbB);

    for (int ks = 0; ks < CC / 32; ++ks) {
        int kn = (ks + 1 < CC / 32) ? (ks + 1) : ks;
        bf16x16 an = load_opA(arow + kn * 32 + kbA);
        bf16x16 bn[4];
#pragma unroll
        for (int s = 0; s < 4; ++s) bn[s] = load_opB(brow[s] + kn * 32 + kbB);
#pragma unroll
        for (int s = 0; s < 4; ++s) acc[s] = wmma_bf16(a, bm[s], acc[s]);
        a = an;
#pragma unroll
        for (int s = 0; s < 4; ++s) bm[s] = bn[s];
    }
#pragma unroll
    for (int s = 0; s < 4; ++s)
#pragma unroll
        for (int j = 0; j < 8; ++j) {
            int r = mt * 16 + j + hi8;
            int c = nt * 64 + s * 16 + (lane & 15);
            float v = acc[s][j] + bias[c];
            int b = r / NN, n = r % NN;
            int which = c / CC, rem = c % CC, h = rem / HD, d = rem % HD;
            size_t bh = (size_t)(b * HH + h);
            if (which == 0)      qb[(bh * NP + n) * HD + d] = (__bf16)(v * 0.125f);
            else if (which == 1) kbuf[(bh * NP + n) * HD + d] = (__bf16)v;
            else                 vtb[(bh * HD + d) * KP + n] = (__bf16)v;
        }
}

// ---------------------------------------------------------------------------
// K2: per-(b,h): S = q @ k^T (WMMA into 208x208 f32 LDS), then per-row
// softmax + 99th-largest threshold (bisection, wave32 shuffle reductions),
// write thresholded bf16 attention ([bh, NP, KP], zero-padded cols) + diag.
// ---------------------------------------------------------------------------
extern __shared__ float Sdyn[];

__global__ void __launch_bounds__(256) attn_kernel(const __bf16* __restrict__ qb,
                                                   const __bf16* __restrict__ kbuf,
                                                   __bf16* __restrict__ attnb,
                                                   float* __restrict__ diagh) {
    int bh = blockIdx.x;
    int lane = threadIdx.x & 31, wave = threadIdx.x >> 5;
    int kbA = (lane & 16) ? 16 : 0;
    int kbB = (lane & 16) ? 16 : 0;
    int hi8 = (lane & 16) ? 8 : 0;
    const __bf16* qbase = qb + (size_t)bh * NP * HD;
    const __bf16* kbase = kbuf + (size_t)bh * NP * HD;

    for (int t = wave; t < 13 * 13; t += 8) {
        int mi = t / 13, ni = t % 13;
        const __bf16* arow = qbase + (size_t)(mi * 16 + (lane & 15)) * HD;
        const __bf16* brow = kbase + (size_t)(ni * 16 + (lane & 15)) * HD;
        // issue all 4 operand loads up front, then both WMMAs
        bf16x16 a0 = load_opA(arow + kbA);
        bf16x16 b0 = load_opB(brow + kbB);
        bf16x16 a1 = load_opA(arow + 32 + kbA);
        bf16x16 b1 = load_opB(brow + 32 + kbB);
        f32x8 acc = {};
        acc = wmma_bf16(a0, b0, acc);
        acc = wmma_bf16(a1, b1, acc);
#pragma unroll
        for (int j = 0; j < 8; ++j)
            Sdyn[(mi * 16 + j + hi8) * NP + ni * 16 + (lane & 15)] = acc[j];
    }
    __syncthreads();

    for (int r = wave; r < NN; r += 8) {
        float* row = Sdyn + (size_t)r * NP;
        float mx = -3.4e38f;
        for (int c = lane; c < NN; c += 32) mx = fmaxf(mx, row[c]);
        mx = wmaxf(mx);
        float sum = 0.f;
        for (int c = lane; c < NN; c += 32) sum += __expf(row[c] - mx);
        sum = wsum(sum);
        float inv = 1.0f / sum;
        for (int c = lane; c < NN; c += 32) row[c] = __expf(row[c] - mx) * inv;
        // lanes only read their own columns below -> no extra sync needed
        float lo = 0.f, hi = 1.0f;
        for (int it = 0; it < 30; ++it) {
            float mid = 0.5f * (lo + hi);
            float cnt = 0.f;
            for (int c = lane; c < NN; c += 32) cnt += (row[c] >= mid) ? 1.f : 0.f;
            cnt = wsum(cnt);
            if (cnt >= 99.f) lo = mid; else hi = mid;   // kth=98 -> keep top >=99
        }
        __bf16* orow = attnb + (size_t)(bh * NP + r) * KP;
        for (int c = lane; c < KP; c += 32) {
            float a = 0.f;
            if (c < NN) { a = row[c]; if (a < lo) a = 0.f; }
            orow[c] = (__bf16)a;
            if (c == r) diagh[(size_t)bh * NP + r] = a;
        }
    }
}

// ---------------------------------------------------------------------------
// K3: per batch: diag mean over heads, stable descending argsort of tokens
// 1..196, build sorted idx_kept (181, incl. cls 0) and idx_prop (16).
// ---------------------------------------------------------------------------
__global__ void __launch_bounds__(256) select_kernel(const float* __restrict__ diagh,
                                                     int* __restrict__ idxk,
                                                     int* __restrict__ idxp) {
    __shared__ float dg[NN];
    __shared__ int order[NN - 1];
    __shared__ unsigned char kf[NN];
    int b = blockIdx.x, tid = threadIdx.x;
    for (int n = tid; n < NN; n += 256) {
        float s = 0.f;
        for (int h = 0; h < HH; ++h) s += diagh[(size_t)(b * HH + h) * NP + n];
        dg[n] = s * (1.0f / HH);
    }
    __syncthreads();
    for (int t = 1 + tid; t < NN; t += 256) {
        float key = dg[t];
        int pos = 0;
        for (int j = 1; j < NN; ++j) {
            float dj = dg[j];
            pos += (dj > key) || (dj == key && j < t);
        }
        order[pos] = t;   // token ids 1..196, descending by diag, stable
    }
    __syncthreads();
    for (int i = tid; i < NN; i += 256) kf[i] = 0;
    __syncthreads();
    if (tid == 0) kf[0] = 1;
    for (int p = tid; p < NKEPT - 1; p += 256) kf[order[p]] = 1;
    __syncthreads();
    for (int t = tid; t < NN; t += 256)
        if (kf[t]) {
            int pos = 0;
            for (int j = 0; j < t; ++j) pos += kf[j];
            idxk[b * NKEPT + pos] = t;
        }
    for (int p = tid; p < NPROP; p += 256) {
        int tok = order[NKEPT - 1 + p];
        int pos = 0;
        for (int q = 0; q < NPROP; ++q) pos += (order[NKEPT - 1 + q] < tok);
        idxp[b * NPROP + pos] = tok;
    }
}

// ---------------------------------------------------------------------------
// K4: per-(b,h): norms + 181x16 dot products -> sim, argmax over kept per
// prop (first-max tie break), then in-place merge add 0.1*prop row into the
// chosen kept row of the bf16 attention buffer.
// ---------------------------------------------------------------------------
__global__ void __launch_bounds__(256) merge_kernel(__bf16* __restrict__ attnb,
                                                    const int* __restrict__ idxk,
                                                    const int* __restrict__ idxp) {
    __shared__ int ik[NKEPT];
    __shared__ int ip[NPROP];
    __shared__ float nk[NKEPT];
    __shared__ float np_[NPROP];
    __shared__ float sim[NPROP * NKEPT];
    __shared__ int midx[NPROP];
    int bh = blockIdx.x, b = bh / HH, tid = threadIdx.x;
    for (int i = tid; i < NKEPT; i += 256) ik[i] = idxk[b * NKEPT + i];
    for (int i = tid; i < NPROP; i += 256) ip[i] = idxp[b * NPROP + i];
    __syncthreads();
    __bf16* base = attnb + (size_t)bh * NP * KP;
    for (int i = tid; i < NKEPT + NPROP; i += 256) {
        int row = (i < NKEPT) ? ik[i] : ip[i - NKEPT];
        const __bf16* pr = base + (size_t)row * KP;
        float s = 0.f;
        for (int c = 0; c < NN; ++c) { float v = (float)pr[c]; s += v * v; }
        if (i < NKEPT) nk[i] = s; else np_[i - NKEPT] = s;
    }
    __syncthreads();
    for (int pair = tid; pair < NKEPT * NPROP; pair += 256) {
        int k = pair % NKEPT, p = pair / NKEPT;
        const __bf16* ra = base + (size_t)ik[k] * KP;
        const __bf16* rb = base + (size_t)ip[p] * KP;
        float d = 0.f;
        for (int c = 0; c < NN; ++c) d += (float)ra[c] * (float)rb[c];
        float sq = nk[k] + np_[p] - 2.f * d;
        sim[p * NKEPT + k] = 1.f - sqrtf(fmaxf(sq, 1e-12f));
    }
    __syncthreads();
    if (tid < NPROP) {
        float best = -3.4e38f;
        int bi = 0;
        for (int k = 0; k < NKEPT; ++k) {
            float s = sim[tid * NKEPT + k];
            if (s > best) { best = s; bi = k; }
        }
        midx[tid] = bi;
    }
    __syncthreads();
    // sequential over props; each column owned by one thread -> duplicate
    // targets accumulate correctly without atomics.
    for (int p = 0; p < NPROP; ++p) {
        __bf16* dst = base + (size_t)ik[midx[p]] * KP;
        const __bf16* src = base + (size_t)ip[p] * KP;
        for (int c = tid; c < NN; c += 256)
            dst[c] = (__bf16)((float)dst[c] + 0.1f * (float)src[c]);
    }
}

// ---------------------------------------------------------------------------
// K5: per-(b,h): out = attn_kept(181 x 224) @ v via WMMA with indirect
// (gathered) A rows; software-pipelined over the 7 K-steps; writes bf16
// [b*181+m, h*64+d] for the projection GEMM.
// ---------------------------------------------------------------------------
__global__ void __launch_bounds__(256) av_gemm_kernel(const __bf16* __restrict__ attnb,
                                                      const __bf16* __restrict__ vtb,
                                                      const int* __restrict__ idxk,
                                                      __bf16* __restrict__ outmid) {
    int bh = blockIdx.x, b = bh / HH, h = bh % HH;
    int lane = threadIdx.x & 31, wave = threadIdx.x >> 5;
    int kbA = (lane & 16) ? 16 : 0;
    int kbB = (lane & 16) ? 16 : 0;
    int hi8 = (lane & 16) ? 8 : 0;
    const __bf16* abase = attnb + (size_t)bh * NP * KP;
    const __bf16* vbase = vtb + (size_t)bh * HD * KP;
    for (int t = wave; t < 48; t += 8) {   // 12 m-tiles x 4 n-tiles
        int mt = t >> 2, nt = t & 3;
        int m = mt * 16 + (lane & 15);
        int tok = idxk[b * NKEPT + ((m < NKEPT) ? m : (NKEPT - 1))];
        const __bf16* arow = abase + (size_t)tok * KP;
        const __bf16* brow = vbase + (size_t)(nt * 16 + (lane & 15)) * KP;
        f32x8 acc = {};
        bf16x16 a = load_opA(arow + kbA);
        bf16x16 bm = load_opB(brow + kbB);
        for (int ks = 0; ks < KP / 32; ++ks) {
            int kn = (ks + 1 < KP / 32) ? (ks + 1) : ks;
            bf16x16 an = load_opA(arow + kn * 32 + kbA);
            bf16x16 bn = load_opB(brow + kn * 32 + kbB);
            acc = wmma_bf16(a, bm, acc);
            a = an; bm = bn;
        }
#pragma unroll
        for (int j = 0; j < 8; ++j) {
            int m2 = mt * 16 + j + hi8;
            if (m2 < NKEPT)
                outmid[((size_t)(b * NKEPT + m2)) * CC + h * HD + nt * 16 + (lane & 15)] =
                    (__bf16)acc[j];
        }
    }
}

// ---------------------------------------------------------------------------
// K6: projection GEMM [23168 x 768] @ Wproj^T + bias + gathered origin -> f32
// (software-pipelined like K1)
// ---------------------------------------------------------------------------
__global__ void __launch_bounds__(256) proj_gemm_kernel(const __bf16* __restrict__ outmid,
                                                        const __bf16* __restrict__ wpb,
                                                        const float* __restrict__ bias,
                                                        const float* __restrict__ origin,
                                                        const int* __restrict__ idxk,
                                                        float* __restrict__ out) {
    int wid = (blockIdx.x * 256 + threadIdx.x) >> 5;
    if (wid >= MT_PROJ * NT64_PROJ) return;
    int lane = threadIdx.x & 31;
    int mt = wid / NT64_PROJ, nt = wid % NT64_PROJ;
    int kbA = (lane & 16) ? 16 : 0;
    int kbB = (lane & 16) ? 16 : 0;
    int hi8 = (lane & 16) ? 8 : 0;
    const __bf16* arow = outmid + (size_t)(mt * 16 + (lane & 15)) * CC;
    const __bf16* brow[4];
#pragma unroll
    for (int s = 0; s < 4; ++s)
        brow[s] = wpb + (size_t)(nt * 64 + s * 16 + (lane & 15)) * CC;

    f32x8 acc[4] = {{}, {}, {}, {}};
    bf16x16 a = load_opA(arow + kbA);
    bf16x16 bm[4];
#pragma unroll
    for (int s = 0; s < 4; ++s) bm[s] = load_opB(brow[s] + kbB);

    for (int ks = 0; ks < CC / 32; ++ks) {
        int kn = (ks + 1 < CC / 32) ? (ks + 1) : ks;
        bf16x16 an = load_opA(arow + kn * 32 + kbA);
        bf16x16 bn[4];
#pragma unroll
        for (int s = 0; s < 4; ++s) bn[s] = load_opB(brow[s] + kn * 32 + kbB);
#pragma unroll
        for (int s = 0; s < 4; ++s) acc[s] = wmma_bf16(a, bm[s], acc[s]);
        a = an;
#pragma unroll
        for (int s = 0; s < 4; ++s) bm[s] = bn[s];
    }
#pragma unroll
    for (int s = 0; s < 4; ++s)
#pragma unroll
        for (int j = 0; j < 8; ++j) {
            int r = mt * 16 + j + hi8;
            int c = nt * 64 + s * 16 + (lane & 15);
            int b = r / NKEPT, m = r % NKEPT;
            int tok = idxk[b * NKEPT + m];
            out[(size_t)r * CC + c] =
                acc[s][j] + bias[c] + origin[((size_t)b * NN + tok) * CC + c];
        }
}

// ---------------------------------------------------------------------------
// Host launcher
// ---------------------------------------------------------------------------
extern "C" void kernel_launch(void* const* d_in, const int* in_sizes, int n_in,
                              void* d_out, int out_size, void* d_ws, size_t ws_size,
                              hipStream_t stream) {
    const float* x      = (const float*)d_in[0];
    const float* origin = (const float*)d_in[1];
    const float* Wqkv   = (const float*)d_in[2];
    const float* bqkv   = (const float*)d_in[3];
    const float* Wproj  = (const float*)d_in[4];
    const float* bproj  = (const float*)d_in[5];
    // num_prop (d_in[6]) is fixed at 16 per setup_inputs.

    char* ws = (char*)d_ws;
    size_t off = 0;
    auto carve = [&](size_t bytes) -> char* {
        char* p = ws + off;
        off += (bytes + 255) & ~(size_t)255;
        return p;
    };

    const size_t nBH = (size_t)BB * HH;
    __bf16* xb     = (__bf16*)carve((size_t)ROWS * CC * 2);
    __bf16* wqkvb  = (__bf16*)carve((size_t)3 * CC * CC * 2);
    __bf16* wprojb = (__bf16*)carve((size_t)CC * CC * 2);
    __bf16* qb     = (__bf16*)carve(nBH * NP * HD * 2);
    __bf16* kb     = (__bf16*)carve(nBH * NP * HD * 2);
    __bf16* vtb    = (__bf16*)carve(nBH * HD * KP * 2);
    __bf16* attnb  = (__bf16*)carve(nBH * NP * KP * 2);
    float*  diagh  = (float*)carve(nBH * NP * 4);
    int*    idxk   = (int*)carve((size_t)BB * NKEPT * 4);
    int*    idxp   = (int*)carve((size_t)BB * NPROP * 4);
    __bf16* outmid = (__bf16*)carve((size_t)ROWS_P * CC * 2);

    // zero the padded q/k/vT buffers so padded rows/cols are exact zeros
    hipMemsetAsync(qb, 0, nBH * NP * HD * 2, stream);
    hipMemsetAsync(kb, 0, nBH * NP * HD * 2, stream);
    hipMemsetAsync(vtb, 0, nBH * HD * KP * 2, stream);

    // K0: casts
    {
        int n = ROWS * CC;
        cast_bf16_kernel<<<(n + 255) / 256, 256, 0, stream>>>(x, xb, n);
        n = 3 * CC * CC;
        cast_bf16_kernel<<<(n + 255) / 256, 256, 0, stream>>>(Wqkv, wqkvb, n);
        n = CC * CC;
        cast_bf16_kernel<<<(n + 255) / 256, 256, 0, stream>>>(Wproj, wprojb, n);
    }

    // K1: QKV GEMM
    {
        int waves = MT_QKV * NT64_QKV;
        qkv_gemm_kernel<<<(waves + 7) / 8, 256, 0, stream>>>(xb, wqkvb, bqkv, qb, kb, vtb);
    }

    // K2: scores + softmax + threshold (S lives in 208x208 f32 dynamic LDS)
    attn_kernel<<<(int)nBH, 256, NP * NP * sizeof(float), stream>>>(qb, kb, attnb, diagh);

    // K3: per-batch token ranking + index build
    select_kernel<<<BB, 256, 0, stream>>>(diagh, idxk, idxp);

    // K4: similarity + merge (in place on bf16 attention buffer)
    merge_kernel<<<(int)nBH, 256, 0, stream>>>(attnb, idxk, idxp);

    // K5: attn_kept @ v
    av_gemm_kernel<<<(int)nBH, 256, 0, stream>>>(attnb, vtb, idxk, outmid);

    // K6: projection + bias + origin gather -> f32 output
    {
        int waves = MT_PROJ * NT64_PROJ;
        proj_gemm_kernel<<<(waves + 7) / 8, 256, 0, stream>>>(outmid, wprojb, bproj,
                                                              origin, idxk, (float*)d_out);
    }
}
